// AGEnergyLoss_PixelsAssignment_84069689852303
// MI455X (gfx1250) — matrix operand order repaired and checked by reading, not compiled, not run on hardware
//
#include <hip/hip_runtime.h>
#include <hip/hip_bf16.h>

typedef __attribute__((ext_vector_type(16))) _Float16 v16h;
typedef __attribute__((ext_vector_type(8)))  _Float16 v8h;
typedef __attribute__((ext_vector_type(8)))  float    v8f;

namespace {

constexpr int NB = 4, KC = 4, CF = 256;
constexpr int HF = 64, WF = 64, SF = HF * WF;     // 4096 source pixels
constexpr int HP = 256, WP = 256, SP = HP * WP;   // 65536 output pixels
constexpr int RAD = 3, DIAM = 7;
constexpr int NCHUNK = CF / 32;                   // 8 WMMA K-chunks

// ---- workspace layout (float elements) ----
constexpr size_t WS_COUNTS = 0;                         // 16
constexpr size_t WS_PROTO  = 64;                        // 4096: [(k*NB+n)*CF + c]
constexpr size_t WS_G      = 4352;                      // NB*KC*SF
constexpr size_t WS_G0     = WS_G  + (size_t)NB * KC * SF;
constexpr size_t WS_H      = WS_G0 + (size_t)NB * SF;
constexpr size_t WS_V      = WS_H  + (size_t)NB * SF;
constexpr size_t WS_D1     = WS_V  + (size_t)NB * SF;
constexpr size_t WS_D2     = WS_D1 + (size_t)NB * SF;
constexpr size_t WS_YSM    = WS_D2 + (size_t)NB * SF;       // NB*KC*SP
constexpr size_t WS_PAFF   = WS_YSM + (size_t)NB * KC * SP; // 3*1024 partials
constexpr size_t WS_PCRF   = WS_PAFF + 3 * 1024;            // 3*1024 partials
// f16 regions (expressed as float-offsets into ws; cast to _Float16*)
constexpr size_t WS_FT_F   = WS_PCRF + 3 * 1024;            // NB*SF*CF halfs
constexpr size_t WS_BP_F   = WS_FT_F + (size_t)NB * SF * CF / 2; // NB*8*32*16 halfs

__device__ __forceinline__ void softmax4(const float* x, float* o) {
  float m = fmaxf(fmaxf(x[0], x[1]), fmaxf(x[2], x[3]));
  float s = 0.f;
#pragma unroll
  for (int k = 0; k < 4; k++) { o[k] = expf(x[k] - m); s += o[k]; }
  float inv = 1.f / s;
#pragma unroll
  for (int k = 0; k < 4; k++) o[k] *= inv;
}

// ---- 1) class histogram of nearest-downsampled labels -> counts ----
__global__ void k_counts(const int* __restrict__ sup, float* __restrict__ ws) {
  __shared__ int hist[KC];
  int n = blockIdx.x, tid = threadIdx.x;
  if (tid < KC) hist[tid] = 0;
  __syncthreads();
  for (int i = 0; i < SF / 256; i++) {
    int s = tid + i * 256;
    int lbl = sup[(size_t)n * SP + (s >> 6) * (4 * WP) + (s & 63) * 4];
    atomicAdd(&hist[lbl], 1);
  }
  __syncthreads();
  if (tid < KC) ws[WS_COUNTS + n * KC + tid] = fmaxf((float)hist[tid], 1e-4f);
}

// ---- 2) prototype masked spatial mean: proto[k][n][c] ----
__global__ void k_proto(const float* __restrict__ feat, const int* __restrict__ sup,
                        float* __restrict__ ws) {
  int c = blockIdx.x, n = blockIdx.y, tid = threadIdx.x;  // 64 threads
  float acc[KC] = {0.f, 0.f, 0.f, 0.f};
  const float* fc = feat + ((size_t)n * CF + c) * SF;
  for (int i = 0; i < SF / 64; i++) {
    int s = tid + i * 64;
    int lbl = sup[(size_t)n * SP + (s >> 6) * (4 * WP) + (s & 63) * 4];
    acc[lbl] += fc[s];
  }
  __shared__ float red[KC][64];
#pragma unroll
  for (int k = 0; k < KC; k++) red[k][tid] = acc[k];
  __syncthreads();
  for (int off = 32; off > 0; off >>= 1) {
    if (tid < off)
#pragma unroll
      for (int k = 0; k < KC; k++) red[k][tid] += red[k][tid + off];
    __syncthreads();
  }
  if (tid < KC)
    ws[WS_PROTO + ((size_t)tid * NB + n) * CF + c] =
        red[tid][0] / ws[WS_COUNTS + n * KC + tid];
}

// ---- 3) L2-normalize each prototype over CF ----
__global__ void k_protonorm(float* __restrict__ ws) {
  int kn = blockIdx.x, tid = threadIdx.x;  // 256 threads == CF
  size_t base = WS_PROTO + (size_t)kn * CF;
  float v = ws[base + tid];
  __shared__ float red[256];
  red[tid] = v * v;
  __syncthreads();
  for (int off = 128; off > 0; off >>= 1) {
    if (tid < off) red[tid] += red[tid + off];
    __syncthreads();
  }
  float nrm = sqrtf(red[0]);
  ws[base + tid] = v / fmaxf(nrm, 1e-12f);
}

// ---- 4a) pack A: transpose feature (channel-major f32) -> featT f16 [n][s][c]
__global__ void k_pack_a(const float* __restrict__ feat, float* __restrict__ ws) {
  _Float16* ft = (_Float16*)(ws + WS_FT_F);
  __shared__ _Float16 lds[32][65];
  int tid = threadIdx.x;                 // 256
  int s0 = blockIdx.x * 64;              // pixel tile
  int c0 = blockIdx.y * 32;              // channel tile
  int n  = blockIdx.z;
#pragma unroll
  for (int i = 0; i < 8; i++) {          // load 32ch x 64px, coalesced on px
    int e = tid + i * 256;
    int c = e >> 6, p = e & 63;
    lds[c][p] = (_Float16)feat[((size_t)n * CF + c0 + c) * SF + s0 + p];
  }
  __syncthreads();
#pragma unroll
  for (int i = 0; i < 8; i++) {          // store 64px x 32ch, coalesced on ch
    int e = tid + i * 256;
    int s = e >> 5, c = e & 31;
    ft[((size_t)n * SF + s0 + s) * CF + c0 + c] = lds[c][s];
  }
}

// ---- 4b) pack B: per-lane WMMA B register image (classes 0..3, zero pad) ----
__global__ void k_prep_b(float* __restrict__ ws) {
  const float* proto = ws + WS_PROTO;
  _Float16* bp = (_Float16*)(ws + WS_BP_F);
  int lane = threadIdx.x, n = blockIdx.x;
  int m = lane & 15;                     // B column (class)
  int kb16 = (lane >> 4) * 16;           // B-layout K base
  for (int cc = 0; cc < NCHUNK; cc++) {
    int cb = cc * 32;
#pragma unroll
    for (int i = 0; i < 16; i++) {
      _Float16 v = (m < KC)
          ? (_Float16)proto[((size_t)m * NB + n) * CF + cb + kb16 + i]
          : (_Float16)0.0f;
      bp[(((size_t)n * NCHUNK + cc) * 32 + lane) * 16 + i] = v;
    }
  }
}

// ---- 4c) WMMA projection: G[n][k][s] = sum_c featT[n][s][c] * proto_l2[k][n][c]
// One wave per 16-pixel tile; 8 unrolled v_wmma_f32_16x16x32_f16, b128 operand
// loads, no LDS, no barriers.
__global__ void k_gproj(float* __restrict__ ws) {
  const _Float16* ft = (const _Float16*)(ws + WS_FT_F);
  const _Float16* bp = (const _Float16*)(ws + WS_BP_F);
  float* G = ws + WS_G;
  int lane = threadIdx.x;                // 0..31 (wave32)
  int n = blockIdx.y;
  int s0 = blockIdx.x * 16;
  int m = lane & 15;
  int kb8 = (lane >> 4) * 16;            // A-layout K base
  const _Float16* arow = ft + ((size_t)n * SF + s0 + m) * CF;
  const _Float16* brow = bp + (size_t)n * NCHUNK * 32 * 16 + (size_t)lane * 16;
  v8f c = {};
#pragma unroll
  for (int cc = 0; cc < NCHUNK; cc++) {
    int cb = cc * 32;
    v8h alo = *(const v8h*)(arow + cb + kb8);        // K = kb8+0..7
    v8h ahi = *(const v8h*)(arow + cb + kb8 + 8);    // K = kb8+8..15
    v8h blo = *(const v8h*)(brow + (size_t)cc * 32 * 16);
    v8h bhi = *(const v8h*)(brow + (size_t)cc * 32 * 16 + 8);
    v16h a, b;
#pragma unroll
    for (int i = 0; i < 8; i++) {
      a[i] = alo[i]; a[8 + i] = ahi[i];
      b[i] = blo[i]; b[8 + i] = bhi[i];
    }
    c = __builtin_amdgcn_wmma_f32_16x16x32_f16(false, a, false, b,
                                               (short)0, c, false, false);
  }
  if (m < KC) {                          // D: lanes<16 -> M=j, lanes>=16 -> M=8+j
    int mb = (lane >= 16) ? 8 : 0;
#pragma unroll
    for (int j = 0; j < 8; j++)
      G[((size_t)n * KC + m) * SF + s0 + mb + j] = c[j];
  }
}

// ---- 5) Gram maps for exact ||bilinear(feature)|| (clamped neighbors) ----
__global__ void k_gram(const float* __restrict__ feat, float* __restrict__ ws) {
  int t = blockIdx.x * 256 + threadIdx.x;
  int n = t / SF, s = t % SF;
  int x = s & 63, y = s >> 6;
  int dx = (x < WF - 1) ? 1 : 0, dy = (y < HF - 1) ? WF : 0;
  const float* f = feat + (size_t)n * CF * SF;
  float g0 = 0.f, hh = 0.f, vv = 0.f, d1 = 0.f, d2 = 0.f;
  for (int c = 0; c < CF; c++) {
    const float* fc = f + (size_t)c * SF;
    if (c + 8 < CF) __builtin_prefetch(fc + 8 * SF + s, 0, 1);  // global_prefetch
    float a00 = fc[s], a01 = fc[s + dx], a10 = fc[s + dy], a11 = fc[s + dx + dy];
    g0 += a00 * a00; hh += a00 * a01; vv += a00 * a10;
    d1 += a00 * a11; d2 += a01 * a10;
  }
  size_t o = (size_t)n * SF + s;
  ws[WS_G0 + o] = g0; ws[WS_H + o] = hh; ws[WS_V + o] = vv;
  ws[WS_D1 + o] = d1; ws[WS_D2 + o] = d2;
}

// ---- 6) affinity + y_red + softmaxes + bounder partial sums ----
__global__ void k_affinity(const float* __restrict__ yhat, const int* __restrict__ rois,
                           float* __restrict__ ws, float* __restrict__ yred_out) {
  int t = blockIdx.x * 256 + threadIdx.x;
  int n = t >> 16, p = t & (SP - 1);
  int h = p >> 8, w = p & 255;
  float sx = (w + 0.5f) * 0.25f - 0.5f;
  float sy = (h + 0.5f) * 0.25f - 0.5f;
  float fx = floorf(sx), fy = floorf(sy);
  float wx = sx - fx, wy = sy - fy;
  int ix = (int)fx, iy = (int)fy;
  int x0 = min(max(ix, 0), WF - 1), x1 = min(max(ix + 1, 0), WF - 1);
  int y0 = min(max(iy, 0), HF - 1), y1 = min(max(iy + 1, 0), HF - 1);
  float wa = (1.f - wy) * (1.f - wx), wb = (1.f - wy) * wx;
  float wc = wy * (1.f - wx), wd = wy * wx;
  int s00 = y0 * WF + x0, s01 = y0 * WF + x1, s10 = y1 * WF + x0, s11 = y1 * WF + x1;
  size_t nb = (size_t)n * SF;
  const float* G0 = ws + WS_G0 + nb;
  const float* Hm = ws + WS_H + nb;
  const float* Vm = ws + WS_V + nb;
  const float* D1m = ws + WS_D1 + nb;
  const float* D2m = ws + WS_D2 + nb;
  bool bx = x1 > x0, by = y1 > y0;
  float n00 = G0[s00], n01 = G0[s01], n10 = G0[s10], n11 = G0[s11];
  float p0001 = bx ? Hm[s00] : n00;
  float p1011 = bx ? Hm[s10] : n10;
  float p0010 = by ? Vm[s00] : n00;
  float p0111 = by ? Vm[s01] : n01;
  float p0011 = bx ? (by ? D1m[s00] : Hm[s00]) : (by ? Vm[s00] : n00);
  float p0110 = bx ? (by ? D2m[s00] : Hm[s00]) : (by ? Vm[s00] : n00);
  float norm2 = wa * wa * n00 + wb * wb * n01 + wc * wc * n10 + wd * wd * n11 +
                2.f * (wa * wb * p0001 + wc * wd * p1011 + wa * wc * p0010 +
                       wb * wd * p0111 + wa * wd * p0011 + wb * wc * p0110);
  float inv = 1.f / fmaxf(sqrtf(fmaxf(norm2, 0.f)), 1e-12f);
  float cosk[KC], csum = 0.f;
#pragma unroll
  for (int k = 0; k < KC; k++) {
    const float* Gk = ws + WS_G + ((size_t)n * KC + k) * SF;
    float cu = wa * Gk[s00] + wb * Gk[s01] + wc * Gk[s10] + wd * Gk[s11];
    cosk[k] = fmaxf(cu * inv, 0.f);
    csum += cosk[k];
  }
  float adn = 1.f / fmaxf(csum, 1e-10f);
  float yh[KC], yr[KC];
#pragma unroll
  for (int k = 0; k < KC; k++) {
    yh[k] = yhat[((size_t)n * KC + k) * SP + p];
    yr[k] = cosk[k] * adn * yh[k];
    yred_out[((size_t)n * KC + k) * SP + p] = yr[k];
  }
  float ysm[KC]; softmax4(yr, ysm);
#pragma unroll
  for (int k = 0; k < KC; k++) ws[WS_YSM + ((size_t)n * KC + k) * SP + p] = ysm[k];
  float pp[KC]; softmax4(yh, pp);
  float mx = fmaxf(fmaxf(yr[0], yr[1]), fmaxf(yr[2], yr[3]));
  float res[KC];
#pragma unroll
  for (int k = 0; k < KC; k++) res[k] = (yr[k] == mx) ? yr[k] : 0.f;
  float sl2[KC]; softmax4(res, sl2);
  float r = (float)rois[(size_t)n * SP + p];
  float aI = 0.f, aS = 0.f, aT = 0.f;
#pragma unroll
  for (int k = 0; k < KC; k++) {
    float sc = pp[k] * r, tg = sl2[k] * r;
    aI += sc * tg; aS += sc; aT += tg;
  }
  __shared__ float red[256];
  int tid = threadIdx.x;
  float vals[3] = {aI, aS, aT};
  for (int q = 0; q < 3; q++) {
    __syncthreads();
    red[tid] = vals[q];
    __syncthreads();
    for (int off = 128; off > 0; off >>= 1) {
      if (tid < off) red[tid] += red[tid + off];
      __syncthreads();
    }
    if (tid == 0) ws[WS_PAFF + (size_t)q * 1024 + blockIdx.x] = red[0];
  }
}

// ---- 7) CRF 49-tap pass: async-to-LDS tile staging, kernels, prod, partials
__global__ void k_crf(const float* __restrict__ sample, const int* __restrict__ rois_sup,
                      float* __restrict__ ws, float* __restrict__ prod_out) {
  __shared__ float ls[3 + KC][22][22];  // r,g,b,ysm0..3 (valid region only)
  int tx = threadIdx.x, ty = threadIdx.y;
  int tid = ty * 16 + tx;
  int n = blockIdx.z;
  int h0 = blockIdx.y * 16 - RAD, w0 = blockIdx.x * 16 - RAD;
  // gather valid halo via async global->LDS copies (ASYNCcnt path)
  for (int q = tid; q < 22 * 22; q += 256) {
    int ly = q / 22, lx = q % 22;
    int gh = h0 + ly, gw = w0 + lx;
    bool val = ((unsigned)gh < (unsigned)HP) && ((unsigned)gw < (unsigned)WP);
    if (val) {
      size_t pix = (size_t)gh * WP + gw;
#pragma unroll
      for (int ch = 0; ch < 3; ch++) {
        unsigned lofs = (unsigned)(uintptr_t)(&ls[ch][ly][lx]);
        unsigned long long ga =
            (unsigned long long)(uintptr_t)(sample + ((size_t)n * 3 + ch) * SP + pix);
        asm volatile("global_load_async_to_lds_b32 %0, %1, off"
                     :: "v"(lofs), "v"(ga) : "memory");
      }
#pragma unroll
      for (int k = 0; k < KC; k++) {
        unsigned lofs = (unsigned)(uintptr_t)(&ls[3 + k][ly][lx]);
        unsigned long long ga =
            (unsigned long long)(uintptr_t)(ws + WS_YSM + ((size_t)n * KC + k) * SP + pix);
        asm volatile("global_load_async_to_lds_b32 %0, %1, off"
                     :: "v"(lofs), "v"(ga) : "memory");
      }
    }
  }
  asm volatile("s_wait_asynccnt 0x0" ::: "memory");
  __syncthreads();
  int h = blockIdx.y * 16 + ty, w = blockIdx.x * 16 + tx;
  float rc = ls[0][ty + RAD][tx + RAD];
  float gc = ls[1][ty + RAD][tx + RAD];
  float bc = ls[2][ty + RAD][tx + RAD];
  float yc[KC];
#pragma unroll
  for (int k = 0; k < KC; k++) yc[k] = ls[3 + k][ty + RAD][tx + RAD];
  const float inv36 = 1.f / 36.f;
  float far_xy = ((float)(w * w) + (float)(h * h)) * inv36;  // unfold zero-pad: diff = -center
  float ksum = 0.f, pr[KC] = {0.f, 0.f, 0.f, 0.f};
  for (int dy = 0; dy < DIAM; dy++) {
    for (int dx = 0; dx < DIAM; dx++) {
      if (dy == RAD && dx == RAD) continue;  // center tap zeroed
      int hh = h + dy - RAD, ww = w + dx - RAD;
      bool val = ((unsigned)hh < (unsigned)HP) && ((unsigned)ww < (unsigned)WP);
      float rn = val ? ls[0][ty + dy][tx + dx] : 0.f;
      float gn = val ? ls[1][ty + dy][tx + dx] : 0.f;
      float bn = val ? ls[2][ty + dy][tx + dx] : 0.f;
      float xyt = val ? (float)((dx - RAD) * (dx - RAD) + (dy - RAD) * (dy - RAD)) * inv36
                      : far_xy;
      float dr = (rn - rc) * 10.f;  // /0.1
      float dg = (gn - gc) * 10.f;
      float db = (bn - bc) * 10.f;
      float rgbt = dr * dr + dg * dg + db * db;
      float kern = 0.9f * expf(-0.5f * (xyt + rgbt)) + 0.1f * expf(-0.5f * xyt);
      ksum += kern;
#pragma unroll
      for (int k = 0; k < KC; k++) {
        float yn = val ? ls[3 + k][ty + dy][tx + dx] : 0.f;
        pr[k] += kern * yn;
      }
    }
  }
  float rs = (float)rois_sup[(size_t)n * SP + (size_t)h * WP + w];
  float pl = 0.f;
#pragma unroll
  for (int k = 0; k < KC; k++) {
    prod_out[((size_t)n * KC + k) * SP + (size_t)h * WP + w] = pr[k];
    pl += pr[k] * yc[k];
  }
  pl *= rs;
  __shared__ float red[256];
  int blk = (blockIdx.z * gridDim.y + blockIdx.y) * gridDim.x + blockIdx.x;  // 0..1023
  float vals[3] = {ksum, pl, rs};
  for (int q = 0; q < 3; q++) {
    __syncthreads();
    red[tid] = vals[q];
    __syncthreads();
    for (int off = 128; off > 0; off >>= 1) {
      if (tid < off) red[tid] += red[tid + off];
      __syncthreads();
    }
    if (tid == 0) ws[WS_PCRF + (size_t)q * 1024 + blk] = red[0];
  }
}

// ---- 8) deterministic final reduction of per-block partials ----
__global__ void k_finalize(const float* __restrict__ ws, float* __restrict__ out) {
  __shared__ float red[256];
  __shared__ float sums[6];
  int tid = threadIdx.x;
  for (int q = 0; q < 6; q++) {
    const float* p = ws + ((q < 3) ? (WS_PAFF + (size_t)q * 1024)
                                   : (WS_PCRF + (size_t)(q - 3) * 1024));
    float a = 0.f;
    for (int i = tid; i < 1024; i += 256) a += p[i];
    __syncthreads();
    red[tid] = a;
    __syncthreads();
    for (int off = 128; off > 0; off >>= 1) {
      if (tid < off) red[tid] += red[tid + off];
      __syncthreads();
    }
    if (tid == 0) sums[q] = red[0];
  }
  __syncthreads();
  if (tid == 0) {
    float I = sums[0], S = sums[1], T = sums[2];
    float ksum = sums[3], prodsum = sums[4], denom = sums[5];
    out[0] = fmaxf(ksum - prodsum, 1e-5f) / denom;                  // loss
    out[1] = 1.f - (2.f * I + 1e-5f) / (S + T + 1e-5f);             // bounder_loss
  }
}

}  // namespace

extern "C" void kernel_launch(void* const* d_in, const int* in_sizes, int n_in,
                              void* d_out, int out_size, void* d_ws, size_t ws_size,
                              hipStream_t stream) {
  (void)in_sizes; (void)n_in; (void)out_size; (void)ws_size;
  const float* feature  = (const float*)d_in[0];
  const int*   sup_lab  = (const int*)d_in[1];
  const float* y_hat    = (const float*)d_in[2];
  const float* sample   = (const float*)d_in[4];
  const int*   rois     = (const int*)d_in[7];
  const int*   rois_sup = (const int*)d_in[8];
  float* out = (float*)d_out;
  float* ws  = (float*)d_ws;
  float* yred_out = out + 2;
  float* prod_out = out + 2 + (size_t)NB * KC * SP;

  k_counts   <<<NB, 256, 0, stream>>>(sup_lab, ws);
  k_proto    <<<dim3(CF, NB), 64, 0, stream>>>(feature, sup_lab, ws);
  k_protonorm<<<KC * NB, 256, 0, stream>>>(ws);
  k_pack_a   <<<dim3(SF / 64, CF / 32, NB), 256, 0, stream>>>(feature, ws);
  k_prep_b   <<<NB, 32, 0, stream>>>(ws);
  k_gproj    <<<dim3(SF / 16, NB), 32, 0, stream>>>(ws);
  k_gram     <<<(NB * SF) / 256, 256, 0, stream>>>(feature, ws);
  k_affinity <<<(NB * SP) / 256, 256, 0, stream>>>(y_hat, rois, ws, yred_out);
  k_crf      <<<dim3(WP / 16, HP / 16, NB), dim3(16, 16), 0, stream>>>(sample, rois_sup, ws, prod_out);
  k_finalize <<<1, 256, 0, stream>>>(ws, out);
}
